// DocumentWordContextBertNER_31413390803685
// MI455X (gfx1250) — matrix operand order repaired and checked by reading, not compiled, not run on hardware
//
#include <hip/hip_runtime.h>

// ---------------------------------------------------------------------------
// Shapes
// ---------------------------------------------------------------------------
#define BB      32
#define SS      256
#define DD      768
#define K2      1536          // 2*D
#define N4      3072          // 4*D
#define MTOK    (BB * SS)     // 8192 tokens
#define NDOCS   64
#define DVOCAB  2048
#define NCLS    9

typedef __attribute__((ext_vector_type(16))) __bf16        v16bf;
typedef __attribute__((ext_vector_type(8)))  float         v8f;
typedef __attribute__((ext_vector_type(4)))  unsigned int  v4u;
typedef __attribute__((ext_vector_type(8)))  int           v8i;
typedef __attribute__((ext_vector_type(4)))  int           v4i;

union FragBF {
    v16bf v;
    unsigned int u[8];
};

__device__ __forceinline__ unsigned short f32_to_bf16(float f) {
    union { float f; unsigned int u; } cv;
    cv.f = f;
    unsigned int u = cv.u;
    u += 0x7FFFu + ((u >> 16) & 1u);   // round-to-nearest-even
    return (unsigned short)(u >> 16);
}

__device__ __forceinline__ float sigm(float x)  { return 1.0f / (1.0f + __expf(-x)); }
__device__ __forceinline__ float tanh_a(float x){ return 1.0f - 2.0f / (1.0f + __expf(2.0f * x)); }

// ---------------------------------------------------------------------------
// WMMA fragment loaders (wave32; A/B use contiguous k order, same in both)
//   A (16x32 bf16, MxK): lane l: m=l&15, khalf=(l>>4)*16, K=khalf+2v,+1
//   B (32x16 bf16, KxN): lane l: n=l&15, khalf=(l>>4)*16, K=khalf+2v,+1
//   C/D (16x16 f32):     lane l: n=l&15, m=(l>>4)*8 + v
// ---------------------------------------------------------------------------
__device__ __forceinline__ v16bf load_a_frag(const unsigned short* __restrict__ base,
                                             int row_stride, int m0, int k0, unsigned lane) {
    FragBF fr;
    const int m  = m0 + (int)(lane & 15u);
    const int kb = (int)(lane >> 4) * 16;
    const unsigned short* p = base + m * row_stride + k0;
#pragma unroll
    for (int v = 0; v < 8; ++v) {
        int k = kb + 2 * v;
        fr.u[v] = *(const unsigned int*)(p + k);
    }
    return fr.v;
}

// B from row-major weight (N rows, K cols): B[k][n] = w[n][k]
__device__ __forceinline__ v16bf load_b_frag(const unsigned short* __restrict__ w,
                                             int K, int n0, int k0, unsigned lane) {
    FragBF fr;
    const int n  = n0 + (int)(lane & 15u);
    const int kh = (int)(lane >> 4) * 16;
    const unsigned short* p = w + n * K + k0 + kh;
#pragma unroll
    for (int v = 0; v < 8; ++v)
        fr.u[v] = *(const unsigned int*)(p + 2 * v);
    return fr.v;
}

// B fragment from an LDS-staged tile laid out [128 rows(N)][32 cols(K)] bf16
__device__ __forceinline__ v16bf load_b_frag_lds(const unsigned short* tile,
                                                 int n0, unsigned lane) {
    FragBF fr;
    const int n  = n0 + (int)(lane & 15u);
    const int kh = (int)(lane >> 4) * 16;
    const unsigned short* p = tile + n * 32 + kh;
#pragma unroll
    for (int v = 0; v < 8; ++v)
        fr.u[v] = *(const unsigned int*)(p + 2 * v);
    return fr.v;
}

// ---------------------------------------------------------------------------
// TDM: issue tensor_load_to_lds of a 32(K) x 128(N) bf16 tile of w into LDS.
// D# per CDNA5 ISA ch.8: group0 {count,lds_addr,global_addr,type=2},
// group1 {data_size=2B, tensor_dim0=K2, dim0_stride=K2, tile 32x128}.
// Tracked by TENSORcnt; wave-level op (EXEC ignored), so only wave 0 issues.
// This toolchain's builtin is the 6-arg form (groups 2/3 + extra group + cpol).
// ---------------------------------------------------------------------------
__device__ __forceinline__ void tdm_load_btile(const unsigned short* gsrc,
                                               unsigned lds_byte_addr) {
    unsigned long long ga = (unsigned long long)(uintptr_t)gsrc;
    v4u g0;
    g0[0] = 1u;                                            // count=1, user mode
    g0[1] = lds_byte_addr;                                 // lds_addr
    g0[2] = (unsigned)(ga & 0xFFFFFFFFull);                // global_addr[31:0]
    g0[3] = (unsigned)((ga >> 32) & 0x1FFFFFFull)          // global_addr[56:32]
          | (2u << 30);                                    // type = 2 (image)
    v8i g1;
    g1[0] = 0x00010000;                                    // data_size=1 (2 bytes)
    g1[1] = (int)(((unsigned)K2 & 0xFFFFu) << 16);         // tensor_dim0[15:0]
    g1[2] = (int)((K2 >> 16) | (((unsigned)N4 & 0xFFFFu) << 16)); // dim0 hi | dim1 lo
    g1[3] = (int)((N4 >> 16) | (32u << 16));               // dim1 hi | tile_dim0=32
    g1[4] = 128;                                           // tile_dim1=128, tile_dim2=0
    g1[5] = K2;                                            // tensor_dim0_stride
    g1[6] = 0;
    g1[7] = 0;
    v4i z4 = {};
    v8i z8 = {};
    __builtin_amdgcn_tensor_load_to_lds(g0, g1, z4, z4, z8, 0);
}

// ---------------------------------------------------------------------------
// 1. f32 -> bf16 weight conversion
// ---------------------------------------------------------------------------
__global__ void k_cvt_bf16(const float* __restrict__ src, unsigned short* __restrict__ dst, int n) {
    int i = blockIdx.x * blockDim.x + threadIdx.x;
    if (i < n) dst[i] = f32_to_bf16(src[i]);
}

// ---------------------------------------------------------------------------
// 2. Gather mean_embeddings[doc, tok] + concat with last_hidden -> x (bf16)
// ---------------------------------------------------------------------------
__global__ void k_gather_concat(const float* __restrict__ lhs,
                                const float* __restrict__ memb,
                                const int* __restrict__ tokens,
                                const int* __restrict__ docs,
                                unsigned short* __restrict__ xbf) {
    const int token = blockIdx.x;                 // 0..8191 = b*256+s
    const int b   = token >> 8;
    const int doc = docs[b];
    const int tok = tokens[token];
    const float* s1 = lhs  + token * DD;
    const float* s2 = memb + (doc * DVOCAB + tok) * DD;
    unsigned short* dst = xbf + token * K2;
    for (int d = threadIdx.x; d < DD; d += blockDim.x) {
        dst[d]      = f32_to_bf16(s1[d]);
        dst[DD + d] = f32_to_bf16(s2[d]);
    }
}

// ---------------------------------------------------------------------------
// 3. xg = x @ w_ih^T + b_ih + b_hh   (M=8192, N=3072, K=1536), bf16 WMMA.
//    Block = 8 waves; block tile 128(M) x 128(N); wave tile 16(M) x 128(N).
//    B tiles (32K x 128N, 8 KB) staged in LDS by the Tensor Data Mover,
//    double buffered; wave 0 drives the DMA, TENSORcnt + barrier handoff.
// ---------------------------------------------------------------------------
__global__ __launch_bounds__(256) void k_gemm_xg(const unsigned short* __restrict__ x,
                                                 const unsigned short* __restrict__ w,
                                                 const float* __restrict__ bih,
                                                 const float* __restrict__ bhh,
                                                 float* __restrict__ xg) {
    __shared__ unsigned short btile[2][128 * 32];          // 2 x 8 KB

    const unsigned lane = threadIdx.x & 31u;
    const unsigned wv   = threadIdx.x >> 5;
    const int m0   = blockIdx.y * 128 + (int)wv * 16;
    const int nblk = blockIdx.x * 128;

    const unsigned lds0 = (unsigned)(uintptr_t)(&btile[0][0]);
    const unsigned lds1 = (unsigned)(uintptr_t)(&btile[1][0]);
    const int nk = K2 / 32;                                // 48 k-steps

    if (wv == 0)                                           // prologue DMA
        tdm_load_btile(w + nblk * K2 + 0, lds0);

    v8f acc[8] = {};
    for (int ks = 0; ks < nk; ++ks) {
        const int cur = ks & 1;
        if (wv == 0) {
            if (ks + 1 < nk) {
                tdm_load_btile(w + nblk * K2 + (ks + 1) * 32, cur ? lds0 : lds1);
                __builtin_amdgcn_s_wait_tensorcnt(1);      // older (current) done
            } else {
                __builtin_amdgcn_s_wait_tensorcnt(0);      // last tile done
            }
        }
        __syncthreads();                                   // current tile ready

        const v16bf a = load_a_frag(x, K2, m0, ks * 32, lane);
        v16bf bfr[8];
#pragma unroll
        for (int j = 0; j < 8; ++j)
            bfr[j] = load_b_frag_lds(&btile[cur][0], 16 * j, lane);
#pragma unroll
        for (int j = 0; j < 8; ++j)
            acc[j] = __builtin_amdgcn_wmma_f32_16x16x32_bf16(
                false, a, false, bfr[j], (short)0, acc[j], false, false);

        __syncthreads();                                   // done reading cur
    }

    const int n_loc = (int)(lane & 15u);
    const int mb    = (int)(lane >> 4) * 8;
#pragma unroll
    for (int j = 0; j < 8; ++j) {
        const int nc = nblk + 16 * j + n_loc;
        const float bias = bih[nc] + bhh[nc];
#pragma unroll
        for (int v = 0; v < 8; ++v) {
            const int m = m0 + mb + v;
            xg[m * N4 + nc] = acc[j][v] + bias;
        }
    }
}

// ---------------------------------------------------------------------------
// 4. LSTM scan: one block per direction, 32 waves.
//    Per step: g = xg[:,t,:] + h @ w_hh^T  (M=32, N=3072, K=768) via WMMA,
//    c kept in wave registers for all 256 steps, h goes through one LDS buffer.
// ---------------------------------------------------------------------------
__global__ __launch_bounds__(1024) void k_lstm_scan(const float* __restrict__ xg_f,
                                                    const unsigned short* __restrict__ whh_f,
                                                    float* __restrict__ hout_f,
                                                    const float* __restrict__ xg_b,
                                                    const unsigned short* __restrict__ whh_b,
                                                    float* __restrict__ hout_b) {
    __shared__ unsigned short h_lds[BB * DD];              // 48 KB

    const int dir = blockIdx.x;                            // 0 = fwd, 1 = bwd
    const float*          xg   = dir ? xg_b   : xg_f;
    const unsigned short* whh  = dir ? whh_b  : whh_f;
    float*                hout = dir ? hout_b : hout_f;

    const unsigned lane = threadIdx.x & 31u;
    const unsigned wv   = threadIdx.x >> 5;

    for (int i = threadIdx.x; i < BB * DD; i += blockDim.x) h_lds[i] = 0;
    __syncthreads();

    int tm[3], tn[3];
#pragma unroll
    for (int i = 0; i < 3; ++i) {
        const int tid = (int)wv * 3 + i;                   // 0..95
        tm[i] = tid / 48;                                  // batch half
        tn[i] = tid % 48;                                  // hidden /16
    }

    v8f cfr[3] = {};                                       // cell state, resident
    const int n_loc = (int)(lane & 15u);
    const int mb    = (int)(lane >> 4) * 8;

    for (int step = 0; step < SS; ++step) {
        const int t = dir ? (SS - 1 - step) : step;
        float hv[3][8];

#pragma unroll
        for (int i = 0; i < 3; ++i) {
            v8f acc[4] = {};
            for (int k0 = 0; k0 < DD; k0 += 32) {
                const v16bf a = load_a_frag(h_lds, DD, tm[i] * 16, k0, lane);
                v16bf bfr[4];
#pragma unroll
                for (int g = 0; g < 4; ++g)
                    bfr[g] = load_b_frag(whh, DD, g * DD + tn[i] * 16, k0, lane);
#pragma unroll
                for (int g = 0; g < 4; ++g)
                    acc[g] = __builtin_amdgcn_wmma_f32_16x16x32_bf16(
                        false, a, false, bfr[g], (short)0, acc[g], false, false);
            }
            const int nh = tn[i] * 16 + n_loc;
#pragma unroll
            for (int v = 0; v < 8; ++v) {
                const int m    = tm[i] * 16 + mb + v;
                const int base = (m * SS + t) * N4;
                const float gi = acc[0][v] + xg[base + nh];
                const float gf = acc[1][v] + xg[base + DD + nh];
                const float gg = acc[2][v] + xg[base + 2 * DD + nh];
                const float go = acc[3][v] + xg[base + 3 * DD + nh];
                const float c  = sigm(gf) * cfr[i][v] + sigm(gi) * tanh_a(gg);
                cfr[i][v] = c;
                const float h = sigm(go) * tanh_a(c);
                hv[i][v] = h;
                hout[(m * SS + t) * DD + nh] = h;          // global: no LDS hazard
            }
        }
        __syncthreads();                                   // all reads of old h done
#pragma unroll
        for (int i = 0; i < 3; ++i) {
            const int nh = tn[i] * 16 + n_loc;
#pragma unroll
            for (int v = 0; v < 8; ++v) {
                const int m = tm[i] * 16 + mb + v;
                h_lds[m * DD + nh] = f32_to_bf16(hv[i][v]);
            }
        }
        __syncthreads();                                   // new h visible
    }
}

// ---------------------------------------------------------------------------
// 5. out = [h_f, h_b] @ w_lin^T + b_lin   (N=9, tiny -> VALU)
// ---------------------------------------------------------------------------
__global__ void k_final_linear(const float* __restrict__ hf,
                               const float* __restrict__ hb,
                               const float* __restrict__ wl,
                               const float* __restrict__ bl,
                               float* __restrict__ out) {
    const int gid = blockIdx.x * blockDim.x + threadIdx.x;
    if (gid >= MTOK * NCLS) return;
    const int token = gid / NCLS;
    const int cls   = gid - token * NCLS;
    const float* rf = hf + token * DD;
    const float* rb = hb + token * DD;
    const float* w  = wl + cls * K2;
    float acc = bl[cls];
    for (int k = 0; k < DD; ++k) acc += rf[k] * w[k];
    for (int k = 0; k < DD; ++k) acc += rb[k] * w[DD + k];
    out[gid] = acc;
}

// ---------------------------------------------------------------------------
// Launcher
// ---------------------------------------------------------------------------
extern "C" void kernel_launch(void* const* d_in, const int* in_sizes, int n_in,
                              void* d_out, int out_size, void* d_ws, size_t ws_size,
                              hipStream_t stream) {
    const float* lhs    = (const float*)d_in[0];
    const float* memb   = (const float*)d_in[1];
    const int*   tokens = (const int*)d_in[2];
    const int*   docs   = (const int*)d_in[3];
    const float* w_ih_f = (const float*)d_in[4];
    const float* w_hh_f = (const float*)d_in[5];
    const float* b_ih_f = (const float*)d_in[6];
    const float* b_hh_f = (const float*)d_in[7];
    const float* w_ih_b = (const float*)d_in[8];
    const float* w_hh_b = (const float*)d_in[9];
    const float* b_ih_b = (const float*)d_in[10];
    const float* b_hh_b = (const float*)d_in[11];
    const float* w_lin  = (const float*)d_in[12];
    const float* b_lin  = (const float*)d_in[13];
    float*       out    = (float*)d_out;

    // Workspace layout (bytes), total ~305 MB
    char* ws = (char*)d_ws;
    size_t off = 0;
    unsigned short* xbf     = (unsigned short*)(ws + off); off += (size_t)MTOK * K2 * 2;
    unsigned short* wihf_bf = (unsigned short*)(ws + off); off += (size_t)N4 * K2 * 2;
    unsigned short* wihb_bf = (unsigned short*)(ws + off); off += (size_t)N4 * K2 * 2;
    unsigned short* whhf_bf = (unsigned short*)(ws + off); off += (size_t)N4 * DD * 2;
    unsigned short* whhb_bf = (unsigned short*)(ws + off); off += (size_t)N4 * DD * 2;
    float* xg_f = (float*)(ws + off); off += (size_t)MTOK * N4 * 4;
    float* xg_b = (float*)(ws + off); off += (size_t)MTOK * N4 * 4;
    float* h_f  = (float*)(ws + off); off += (size_t)MTOK * DD * 4;
    float* h_b  = (float*)(ws + off); off += (size_t)MTOK * DD * 4;
    (void)in_sizes; (void)n_in; (void)out_size; (void)ws_size;

    // 1. weights -> bf16
    k_cvt_bf16<<<(N4 * K2 + 255) / 256, 256, 0, stream>>>(w_ih_f, wihf_bf, N4 * K2);
    k_cvt_bf16<<<(N4 * K2 + 255) / 256, 256, 0, stream>>>(w_ih_b, wihb_bf, N4 * K2);
    k_cvt_bf16<<<(N4 * DD + 255) / 256, 256, 0, stream>>>(w_hh_f, whhf_bf, N4 * DD);
    k_cvt_bf16<<<(N4 * DD + 255) / 256, 256, 0, stream>>>(w_hh_b, whhb_bf, N4 * DD);

    // 2. gather + concat
    k_gather_concat<<<MTOK, 256, 0, stream>>>(lhs, memb, tokens, docs, xbf);

    // 3. input-gate projections (WMMA + TDM-staged B tiles)
    dim3 ggrid(N4 / 128, MTOK / 128);
    k_gemm_xg<<<ggrid, 256, 0, stream>>>(xbf, wihf_bf, b_ih_f, b_hh_f, xg_f);
    k_gemm_xg<<<ggrid, 256, 0, stream>>>(xbf, wihb_bf, b_ih_b, b_hh_b, xg_b);

    // 4. bidirectional recurrence (one WGP per direction, WMMA in the scan)
    k_lstm_scan<<<2, 1024, 0, stream>>>(xg_f, whhf_bf, h_f, xg_b, whhb_bf, h_b);

    // 5. classifier head
    k_final_linear<<<(MTOK * NCLS + 255) / 256, 256, 0, stream>>>(h_f, h_b, w_lin, b_lin, out);
}
